// MHSA_27582279975470
// MI455X (gfx1250) — compile-verified, matrix operand order checked by reading, not run on hardware
//
#include <hip/hip_runtime.h>
#include <hip/hip_bf16.h>

// ---------- CDNA5 WMMA types ----------
typedef __attribute__((ext_vector_type(16))) __bf16        v16bf;
typedef __attribute__((ext_vector_type(8)))  float         v8f;
typedef __attribute__((ext_vector_type(4)))  unsigned int  v4u;

union Frag16 { v4u u[2]; v16bf bf; };

__device__ __forceinline__ unsigned short f2bf(float f) {
  unsigned u = __float_as_uint(f);
  u += 0x7FFFu + ((u >> 16) & 1u);        // round-to-nearest-even
  return (unsigned short)(u >> 16);
}

// Load one A/B fragment for v_wmma_*_16x16x32_bf16.
// Per ISA layout: lane holds row/col (lane&15); K elements are two contiguous
// 8-element runs at k = 8*hi and k = 16 + 8*hi  (hi = lane>>4).
__device__ __forceinline__ Frag16 load_frag(const unsigned short* rowp, int hi) {
  Frag16 f;
  f.u[0] = *(const v4u*)(rowp + 8 * hi);
  f.u[1] = *(const v4u*)(rowp + 16 + 8 * hi);
  return f;
}

__device__ __forceinline__ v8f wmma_bf16(const Frag16& a, const Frag16& b, v8f c) {
  return __builtin_amdgcn_wmma_f32_16x16x32_bf16(false, a.bf, false, b.bf,
                                                 (short)0, c, false, false);
}

// CDNA5 async global->LDS copy (16B per lane), tracked by ASYNCcnt.
// VDST = LDS byte offset (low 32 bits of the generic shared address).
__device__ __forceinline__ void async_g2l_b128(unsigned lds_off,
                                               const unsigned short* gptr) {
  asm volatile("global_load_async_to_lds_b128 %0, %1, off"
               :: "v"(lds_off), "v"(gptr)
               : "memory");
}

// ---------- problem constants ----------
#define NH 8
#define HD 32
#define DIM 256
#define BB 4
#define HH 64
#define WW 64
#define NQ 4096          // 64*64
#define NKV 1024         // 32*32

// ---------- pack kernels (fp32 -> bf16, layout changes) ----------

// x[b,c,h,w] (c-major) -> xb[b,n,c] (c contiguous), n = h*64+w
__global__ void pack_x_kernel(const float* __restrict__ x, unsigned short* __restrict__ xb) {
  int idx = blockIdx.x * blockDim.x + threadIdx.x;   // over 4*256*4096 = 2^22
  int n = idx & 4095;
  int c = (idx >> 12) & 255;
  int b = idx >> 20;
  float v = x[(((size_t)(b << 8) | c) << 12) + n];
  xb[((size_t)((b << 12) + n)) * 256 + c] = f2bf(v);
}

// Wq[o,c] -> bf16, same layout (row o, c contiguous == B-fragment friendly)
__global__ void pack_wq_kernel(const float* __restrict__ Wq, unsigned short* __restrict__ Wqb) {
  int i = blockIdx.x * 256 + threadIdx.x;            // 65536
  Wqb[i] = f2bf(Wq[i]);
}

// W[o,c,i,j] (OIHW 2x2) -> Wp[o, q4*256 + c], q4 = i*2+j  (im2col K ordering)
__global__ void pack_wconv_kernel(const float* __restrict__ W, unsigned short* __restrict__ Wp) {
  int i = blockIdx.x * 256 + threadIdx.x;            // 262144
  int o = i >> 10, cc = i & 1023;
  int q4 = cc >> 8, c = cc & 255;
  Wp[i] = f2bf(W[(((size_t)o << 8) | c) * 4 + q4]);
}

// ---------- Q projection: Q[n,o] = sum_c xb[n,c] * Wq[o,c] ----------
// M = B*NQ = 16384 (16/block), N = 256 (32/wave), K = 256.
// Output: Qb[b,h,n,d] bf16 (d contiguous -> A-fragment layout for attention)
__global__ void __launch_bounds__(256) gemm_q_kernel(
    const unsigned short* __restrict__ xb, const unsigned short* __restrict__ Wqb,
    unsigned short* __restrict__ Qb) {
  int tid = threadIdx.x, lane = tid & 31, w = tid >> 5;
  int hi = lane >> 4, l16 = lane & 15;
  int row = blockIdx.x * 16 + l16;
  const unsigned short* ap = xb + (size_t)row * 256;

  v8f acc[2] = {};
  for (int kk = 0; kk < 256; kk += 32) {
    Frag16 a = load_frag(ap + kk, hi);
#pragma unroll
    for (int t = 0; t < 2; ++t) {
      int oc = w * 32 + t * 16 + l16;
      Frag16 bfr = load_frag(Wqb + (size_t)oc * 256 + kk, hi);
      acc[t] = wmma_bf16(a, bfr, acc[t]);
    }
  }
#pragma unroll
  for (int t = 0; t < 2; ++t) {
    int oc = w * 32 + t * 16 + l16;
    int h = oc >> 5, d = oc & 31;
#pragma unroll
    for (int r = 0; r < 8; ++r) {
      int M = blockIdx.x * 16 + r + 8 * hi;
      int b = M >> 12, n = M & 4095;
      Qb[((size_t)(b * NH + h) * NQ + n) * HD + d] = f2bf(acc[t][r]);
    }
  }
}

// ---------- K'/V producer: stride-2 conv as GEMM over im2col (K=1024) ----------
// mode 0: K' = conv_k + (rel_w[yo] + rel_h[xo])  -> Kb[b,h,m,d]  (d contiguous)
// mode 1: V  = conv_v                            -> Vb[b,h,d,m]  (m contiguous)
__global__ void __launch_bounds__(256) gemm_kv_kernel(
    const unsigned short* __restrict__ xb, const unsigned short* __restrict__ Wp,
    const float* __restrict__ rel_h, const float* __restrict__ rel_w,
    unsigned short* __restrict__ outp, int mode) {
  int tid = threadIdx.x, lane = tid & 31, w = tid >> 5;
  int hi = lane >> 4, l16 = lane & 15;
  int g = blockIdx.x * 16 + l16;                 // M row = b*1024 + m
  int b = g >> 10, m = g & 1023;
  int yo = m >> 5, xo = m & 31;

  v8f acc[2] = {};
  for (int kk = 0; kk < 1024; kk += 32) {
    int q4 = kk >> 8, c0 = kk & 255;
    int i = q4 >> 1, j = q4 & 1;
    int n = (2 * yo + i) * WW + (2 * xo + j);
    Frag16 a = load_frag(xb + ((size_t)(b * NQ + n)) * 256 + c0, hi);
#pragma unroll
    for (int t = 0; t < 2; ++t) {
      int oc = w * 32 + t * 16 + l16;
      Frag16 bfr = load_frag(Wp + (size_t)oc * 1024 + kk, hi);
      acc[t] = wmma_bf16(a, bfr, acc[t]);
    }
  }
#pragma unroll
  for (int t = 0; t < 2; ++t) {
    int oc = w * 32 + t * 16 + l16;
    int h = oc >> 5, d = oc & 31;
#pragma unroll
    for (int r = 0; r < 8; ++r) {
      int gr = blockIdx.x * 16 + r + 8 * hi;
      int b2 = gr >> 10, m2 = gr & 1023;
      float v = acc[t][r];
      if (mode == 0) {
        v += rel_w[(size_t)(h * HD + d) * 32 + (m2 >> 5)] +
             rel_h[(size_t)(h * HD + d) * 32 + (m2 & 31)];
        outp[((size_t)(b2 * NH + h) * NKV + m2) * HD + d] = f2bf(v);
      } else {
        outp[((size_t)(b2 * NH + h) * HD + d) * NKV + m2] = f2bf(v);
      }
    }
  }
}

// ---------- fused flash attention ----------
// grid: B*NH*(NQ/64) = 2048 blocks of 128 threads (4 waves).
// Wave w owns 16 q rows; kv processed in chunks of 32, double-buffered in LDS
// via CDNA5 async global->LDS copies (ASYNCcnt).
__global__ void __launch_bounds__(128) attn_kernel(
    const unsigned short* __restrict__ Qb, const unsigned short* __restrict__ Kb,
    const unsigned short* __restrict__ Vb, float* __restrict__ outp) {
  __shared__ __align__(16) unsigned short sK[2][32 * 32];   // K'[m_local, d]
  __shared__ __align__(16) unsigned short sV[2][32 * 32];   // V [d, m_local]
  __shared__ __align__(16) unsigned short sP[4][16 * 32];   // per-wave P transpose

  const float scale = 0.17677669529663687f;                 // 32^-0.5
  int tid = threadIdx.x, lane = tid & 31, w = tid >> 5;
  int hi = lane >> 4, l16 = lane & 15;
  int bid = blockIdx.x;
  int qt = bid & 63, h = (bid >> 6) & 7, b = bid >> 9;
  size_t bh = (size_t)(b * NH + h);
  int n0 = qt * 64 + w * 16;

  // Q fragment: K-dim = full head_dim = 32 -> load once.
  Frag16 aQ = load_frag(Qb + (bh * NQ + n0 + l16) * HD, hi);

  v8f o0 = {}, o1 = {};
  float mrow[8], lrow[8];
#pragma unroll
  for (int r = 0; r < 8; ++r) { mrow[r] = -1e30f; lrow[r] = 0.0f; }

  const unsigned short* Kbase = Kb + bh * NKV * HD;
  const unsigned short* Vbase = Vb + bh * HD * NKV;

  // per-thread cooperative-staging coordinates: 128 threads x 16B per tile
  int rowi = tid >> 2, boff = (tid & 3) * 8;
  unsigned ldsK0 = (unsigned)(unsigned long long)&sK[0][rowi * 32 + boff];
  unsigned ldsK1 = (unsigned)(unsigned long long)&sK[1][rowi * 32 + boff];
  unsigned ldsV0 = (unsigned)(unsigned long long)&sV[0][rowi * 32 + boff];
  unsigned ldsV1 = (unsigned)(unsigned long long)&sV[1][rowi * 32 + boff];

  // prologue: async-stage chunk 0 into buffer 0
  async_g2l_b128(ldsK0, Kbase + (size_t)rowi * HD + boff);
  async_g2l_b128(ldsV0, Vbase + (size_t)rowi * NKV + boff);

  for (int it = 0; it < NKV / 32; ++it) {
    int mc = it * 32;
    int cur = it & 1;
    const unsigned short* cK = &sK[cur][0];
    const unsigned short* cV = &sV[cur][0];

    if (it < NKV / 32 - 1) {
      // issue next chunk into the other buffer (readers of it finished at the
      // tail barrier of the previous iteration)
      int mn = mc + 32;
      async_g2l_b128(cur ? ldsK0 : ldsK1, Kbase + (size_t)(mn + rowi) * HD + boff);
      async_g2l_b128(cur ? ldsV0 : ldsV1, Vbase + (size_t)rowi * NKV + mn + boff);
      // 2 ops outstanding for next chunk; in-order completion => current landed
      asm volatile("s_wait_asynccnt 0x2" ::: "memory");
    } else {
      asm volatile("s_wait_asynccnt 0x0" ::: "memory");
    }
    __syncthreads();   // all waves' current-chunk async copies visible

    // S = Q^T K'  (two 16-wide column tiles, K=32=head_dim, one WMMA each)
    Frag16 bK0 = load_frag(cK + (size_t)l16 * 32, hi);
    Frag16 bK1 = load_frag(cK + (size_t)(16 + l16) * 32, hi);
    v8f z = {};
    v8f s0 = wmma_bf16(aQ, bK0, z);
    v8f s1 = wmma_bf16(aQ, bK1, z);

    // online softmax in C-fragment layout (row = r + 8*hi, col = l16)
    float p0[8], p1[8];
#pragma unroll
    for (int r = 0; r < 8; ++r) {
      float a0 = s0[r] * scale, a1 = s1[r] * scale;
      float v = fmaxf(a0, a1);
      v = fmaxf(v, __shfl_xor(v, 1, 32));
      v = fmaxf(v, __shfl_xor(v, 2, 32));
      v = fmaxf(v, __shfl_xor(v, 4, 32));
      v = fmaxf(v, __shfl_xor(v, 8, 32));
      float mn = fmaxf(mrow[r], v);
      float es = __expf(mrow[r] - mn);
      mrow[r] = mn;
      float e0 = __expf(a0 - mn), e1 = __expf(a1 - mn);
      p0[r] = e0; p1[r] = e1;
      float sum = e0 + e1;
      sum += __shfl_xor(sum, 1, 32);
      sum += __shfl_xor(sum, 2, 32);
      sum += __shfl_xor(sum, 4, 32);
      sum += __shfl_xor(sum, 8, 32);
      lrow[r] = lrow[r] * es + sum;
      o0[r] *= es;
      o1[r] *= es;
    }

    // C-layout -> A-layout transpose of P via per-wave LDS bounce
    unsigned short* pw = &sP[w][0];
#pragma unroll
    for (int r = 0; r < 8; ++r) {
      int rr = r + 8 * hi;
      pw[rr * 32 + l16]      = f2bf(p0[r]);
      pw[rr * 32 + 16 + l16] = f2bf(p1[r]);
    }
    asm volatile("s_wait_dscnt 0x0" ::: "memory");   // in-wave DS RAW fence

    // O += P @ V^T : A = P (16 x 32 kv), B = V^T (32 kv x 16 d), two d-halves
    Frag16 aP  = load_frag(pw + (size_t)l16 * 32, hi);
    Frag16 bV0 = load_frag(cV + (size_t)l16 * 32, hi);
    Frag16 bV1 = load_frag(cV + (size_t)(16 + l16) * 32, hi);
    o0 = wmma_bf16(aP, bV0, o0);
    o1 = wmma_bf16(aP, bV1, o1);

    __syncthreads();   // all readers done before next iter overwrites buf^1
  }

  // normalize and scatter to out[b, h*32+d, y, x]  (n = y*64+x)
#pragma unroll
  for (int r = 0; r < 8; ++r) {
    int n = n0 + r + 8 * hi;
    float inv = 1.0f / lrow[r];
    outp[((size_t)(b * DIM + h * HD + l16)) * NQ + n]      = o0[r] * inv;
    outp[((size_t)(b * DIM + h * HD + 16 + l16)) * NQ + n] = o1[r] * inv;
  }
}

// ---------- host ----------
extern "C" void kernel_launch(void* const* d_in, const int* in_sizes, int n_in,
                              void* d_out, int out_size, void* d_ws, size_t ws_size,
                              hipStream_t stream) {
  const float* x     = (const float*)d_in[0];
  const float* Wq    = (const float*)d_in[1];
  const float* Wk    = (const float*)d_in[2];
  const float* Wv    = (const float*)d_in[3];
  const float* rel_h = (const float*)d_in[4];
  const float* rel_w = (const float*)d_in[5];
  float* out = (float*)d_out;

  char* ws = (char*)d_ws;
  size_t off = 0;
  unsigned short* xb  = (unsigned short*)(ws + off); off += (size_t)BB * NQ * 256 * 2;   // 8 MB
  unsigned short* Wqb = (unsigned short*)(ws + off); off += (size_t)256 * 256 * 2;       // 128 KB
  unsigned short* Wkp = (unsigned short*)(ws + off); off += (size_t)256 * 1024 * 2;      // 512 KB
  unsigned short* Wvp = (unsigned short*)(ws + off); off += (size_t)256 * 1024 * 2;      // 512 KB
  unsigned short* Qb  = (unsigned short*)(ws + off); off += (size_t)BB * NH * NQ * HD * 2;   // 8 MB
  unsigned short* Kb  = (unsigned short*)(ws + off); off += (size_t)BB * NH * NKV * HD * 2;  // 2 MB
  unsigned short* Vb  = (unsigned short*)(ws + off); off += (size_t)BB * NH * NKV * HD * 2;  // 2 MB

  // 1) packs
  pack_x_kernel<<<(BB * 256 * NQ) / 256, 256, 0, stream>>>(x, xb);
  pack_wq_kernel<<<(256 * 256) / 256, 256, 0, stream>>>(Wq, Wqb);
  pack_wconv_kernel<<<(256 * 1024) / 256, 256, 0, stream>>>(Wk, Wkp);
  pack_wconv_kernel<<<(256 * 1024) / 256, 256, 0, stream>>>(Wv, Wvp);

  // 2) projections (WMMA GEMMs)
  gemm_q_kernel<<<(BB * NQ) / 16, 256, 0, stream>>>(xb, Wqb, Qb);
  gemm_kv_kernel<<<(BB * NKV) / 16, 256, 0, stream>>>(xb, Wkp, rel_h, rel_w, Kb, 0);
  gemm_kv_kernel<<<(BB * NKV) / 16, 256, 0, stream>>>(xb, Wvp, rel_h, rel_w, Vb, 1);

  // 3) fused flash attention (async double-buffered K/V staging)
  attn_kernel<<<BB * NH * (NQ / 64), 128, 0, stream>>>(Qb, Kb, Vb, out);
}
